// CNN_symmetric_9723805958629
// MI455X (gfx1250) — compile-verified, matrix operand order checked by reading, not run on hardware
//
#include <hip/hip_runtime.h>
#include <hip/hip_bf16.h>

#define L_    16
#define NC_   256
#define NS_   768
#define NROT_ 8
#define NF_   12
#define B_    128
#define KTAP_ 9
#define BR_   (B_ * NROT_)   // 1024 rows

typedef __attribute__((ext_vector_type(16))) _Float16 v16h;
typedef __attribute__((ext_vector_type(8)))  _Float16 v8h;
typedef __attribute__((ext_vector_type(8)))  float    v8f;

// ---------------------------------------------------------------------------
// workspace layout (bytes)
// ---------------------------------------------------------------------------
#define PLANE_HALVES  ((size_t)BR_ * NC_ * 16)        // one f16 plane, ch stride 16
#define PLANE_BYTES   (PLANE_HALVES * 2)              // 8 MB
#define OFF_AR   ((size_t)0)
#define OFF_AI   (1 * PLANE_BYTES)
#define OFF_BR   (2 * PLANE_BYTES)
#define OFF_BI   (3 * PLANE_BYTES)
#define OFF_CR   (4 * PLANE_BYTES)
#define OFF_CI   (5 * PLANE_BYTES)
#define OFF_Y    (6 * PLANE_BYTES)                    // complex f32 [BR][NC][NF]
#define Y_BYTES  ((size_t)BR_ * NC_ * NF_ * 2 * 4)
#define OFF_BF   (OFF_Y + Y_BYTES)                    // 6 weight-fragment blocks
#define BFRAG_HALVES  ((size_t)7 * 2 * 32 * 16)       // max nk=7
#define BF_BYTES      (6 * BFRAG_HALVES * 2)
#define OFF_OUTA (OFF_BF + BF_BYTES)                  // float2 per row

// ---------------------------------------------------------------------------
__device__ __forceinline__ float block_reduce(float v, float* rbuf, int tid) {
    rbuf[tid] = v;
    __syncthreads();
    for (int st = 128; st > 0; st >>= 1) {
        if (tid < st) rbuf[tid] += rbuf[tid + st];
        __syncthreads();
    }
    float r = rbuf[0];
    __syncthreads();
    return r;
}

// ---------------------------------------------------------------------------
// Kernel 1: symmetrize, triangles (u1), GF(2) phase2 (u0), alpha terms
// one block per row r = b*8 + rot, 256 threads
// ---------------------------------------------------------------------------
__global__ __launch_bounds__(256)
void prep_kernel(const int* __restrict__ x,   const int* __restrict__ pg,
                 const int* __restrict__ tsite, const int* __restrict__ tcell,
                 const int* __restrict__ invm,  const int* __restrict__ tfm,
                 const int* __restrict__ ltri,  const int* __restrict__ rtri,
                 const float* __restrict__ kx,  const float* __restrict__ ky,
                 const float* __restrict__ alpha0, const float* __restrict__ alpha1,
                 _Float16* Ar, _Float16* Ai, _Float16* Cr, _Float16* Ci,
                 float* outA)
{
    const int r   = blockIdx.x;
    const int b   = r >> 3, rot = r & 7;
    const int tid = threadIdx.x;

    __shared__ short sx[NS_];
    __shared__ short sz[NS_];
    __shared__ short sres[NS_];
    __shared__ short su[NC_];
    __shared__ float rbuf[256];
    __shared__ float ph[4];
    __shared__ int   sh_dis, sh_shift;

    // xr = x[b, point_group[rot]]
    for (int s = tid; s < NS_; s += 256)
        sx[s] = (short)x[b * NS_ + pg[rot * NS_ + s]];
    __syncthreads();

    // --- u1 triangle products + alpha1 partial, write branch-2 input planes
    float a1r, a1i;
    {
        const int c = tid;   // NC_ == blockDim
        int lp = sx[ltri[c * 3]] * sx[ltri[c * 3 + 1]] * sx[ltri[c * 3 + 2]];
        int rp = sx[rtri[c * 3]] * sx[rtri[c * 3 + 1]] * sx[rtri[c * 3 + 2]];
        size_t base = ((size_t)r * NC_ + c) * 16;
        Cr[base + 0] = (_Float16)(float)lp;  Cr[base + 1] = (_Float16)(float)rp;
        Ci[base + 0] = (_Float16)0.f;        Ci[base + 1] = (_Float16)0.f;
        a1r = alpha1[0] * lp + alpha1[2] * rp;
        a1i = alpha1[1] * lp + alpha1[3] * rp;
    }

    // --- phase sums for state reposition
    float pxr = 0.f, pxi = 0.f, pyr = 0.f, pyi = 0.f;
    for (int s = tid; s < NS_; s += 256) {
        float s2 = 0.5f * (1.0f + (float)sx[s]);
        pxr += kx[2 * s] * s2;  pxi += kx[2 * s + 1] * s2;
        pyr += ky[2 * s] * s2;  pyi += ky[2 * s + 1] * s2;
    }
    float t0 = block_reduce(pxr, rbuf, tid); if (tid == 0) ph[0] = t0;
    float t1 = block_reduce(pxi, rbuf, tid); if (tid == 0) ph[1] = t1;
    float t2 = block_reduce(pyr, rbuf, tid); if (tid == 0) ph[2] = t2;
    float t3 = block_reduce(pyi, rbuf, tid); if (tid == 0) ph[3] = t3;
    __syncthreads();

    if (tid == 0) {
        const float c2pi = (float)L_ / 6.2831853071795864769f;
        float vx = atan2f(ph[1], ph[0]) * c2pi;
        float vy = atan2f(ph[3], ph[2]) * c2pi;
        vx = roundf(vx * 100000.f) * 1e-5f;
        vy = roundf(vy * 100000.f) * 1e-5f;
        int xsh = (int)(vx <= 0.f ? ((float)L_ - ceilf(-vx)) : -ceilf(-vx));
        int ysh = (int)(vy <= 0.f ? ((float)L_ - ceilf(-vy)) : -ceilf(-vy));
        xsh = ((xsh % L_) + L_) % L_;
        ysh = ((ysh % L_) + L_) % L_;
        sh_dis   = ysh * L_ + xsh;
        sh_shift = ((L_ - ysh) % L_) * L_ + ((L_ - xsh) % L_);
    }
    __syncthreads();

    // z = (1 - x_translated)/2
    for (int s = tid; s < NS_; s += 256)
        sz[s] = (short)((1 - (int)sx[tsite[sh_dis * NS_ + s]]) >> 1);
    __syncthreads();

    // u = z @ invm^T mod 2
    {
        int acc = 0;
        const int* rowp = invm + tid * NS_;
        for (int s = 0; s < NS_; ++s) acc += (int)sz[s] * rowp[s];
        su[tid] = (short)(acc & 1);
    }
    __syncthreads();

    // res = (z + u @ tfm^T) mod 2
    for (int s = tid; s < NS_; s += 256) {
        int acc = 0;
        const int* rowp = tfm + s * NC_;
        for (int c = 0; c < NC_; ++c) acc += (int)su[c] * rowp[c];
        sres[s] = (short)(((int)sz[s] + acc) & 1);
    }
    __syncthreads();

    // a = res @ tfm ; u = (u + (a>3)) mod 2
    {
        int acc = 0;
        for (int s = 0; s < NS_; ++s) acc += (int)sres[s] * tfm[s * NC_ + tid];
        __syncthreads();
        su[tid] = (short)(((int)su[tid] + (acc > 3 ? 1 : 0)) & 1);
    }
    __syncthreads();

    // res recomputed with updated u
    for (int s = tid; s < NS_; s += 256) {
        int acc = 0;
        const int* rowp = tfm + s * NC_;
        for (int c = 0; c < NC_; ++c) acc += (int)su[c] * rowp[c];
        sres[s] = (short)(((int)sz[s] + acc) & 1);
    }
    __syncthreads();

    // gather back by shift, emit u0 planes + alpha0 partial
    float a0r = 0.f, a0i = 0.f;
    {
        const int cell = tid;
        int u0v[4];
        u0v[0] = su[tcell[sh_shift * NC_ + cell]];
        #pragma unroll
        for (int k = 0; k < 3; ++k)
            u0v[1 + k] = sres[tsite[sh_shift * NS_ + 3 * cell + k]];
        size_t base = ((size_t)r * NC_ + cell) * 16;
        #pragma unroll
        for (int k = 0; k < 4; ++k) {
            Ar[base + k] = (_Float16)(float)u0v[k];
            Ai[base + k] = (_Float16)0.f;
            a0r += alpha0[2 * k]     * u0v[k];
            a0i += alpha0[2 * k + 1] * u0v[k];
        }
    }
    float sr = block_reduce(a0r + a1r, rbuf, tid);
    float si = block_reduce(a0i + a1i, rbuf, tid);
    if (tid == 0) { outA[2 * r] = sr; outA[2 * r + 1] = si; }
}

// ---------------------------------------------------------------------------
// Kernel 2: expand complex weights into per-lane WMMA B-fragments.
// K order (padded): kappa = dy*KP + dx*(2*CIN) + p*CIN + c,
//   KP = per-dy span (6*CIN) rounded up to multiple of 8; pad rows are zero.
// B_re[kappa,f] = p ? -Wi : Wr ; B_im[kappa,f] = p ? Wr : Wi
// ---------------------------------------------------------------------------
template <int CIN, int NK>
__global__ void wprep_kernel(const float* __restrict__ W, _Float16* __restrict__ out)
{
    constexpr int CSTR  = 2 * CIN;
    constexpr int KSPAN = 3 * CSTR;
    constexpr int KP    = (KSPAN + 7) & ~7;
    const int l = threadIdx.x;                 // 32 lanes
    const int n = l & 15, khalf = l >> 4;

    #pragma unroll
    for (int s = 0; s < NK; ++s) {
        #pragma unroll
        for (int i = 0; i < 16; ++i) {
            int krel = (i < 8) ? (i + 8 * khalf) : (i + 8 + 8 * khalf);
            int k = s * 32 + krel;
            float vre = 0.f, vim = 0.f;
            int dy = k / KP, kp = k - dy * KP;
            if (dy < 3 && kp < KSPAN && n < NF_) {
                int dx  = kp / CSTR;
                int rem = kp - dx * CSTR;
                int p   = rem / CIN;
                int c   = rem - p * CIN;
                int t   = dy * 3 + dx;
                float wr = W[((t * CIN + c) * NF_ + n) * 2];
                float wi = W[((t * CIN + c) * NF_ + n) * 2 + 1];
                vre = p ? -wi : wr;
                vim = p ?  wr : wi;
            }
            out[((s * 2 + 0) * 32 + l) * 16 + i] = (_Float16)vre;
            out[((s * 2 + 1) * 32 + l) * 16 + i] = (_Float16)vim;
        }
    }
}

// ---------------------------------------------------------------------------
// Kernel 3: one conv layer via WMMA (templated: everything constant-folds).
// grid = (64 batch-tiles, 16 lattice rows), 256 threads (8 waves).
// LDS layout sA[m][dy 0..2][x 0..19][p*CIN+c]: x>=16 duplicates x-16, so the
// im2col K-run for any (s,khalf) is 8 contiguous aligned halves -> b128 loads.
// MODE 0: act2 -> f16 planes ; 1: write complex f32 Y ; 2: accumulate into Y
// ---------------------------------------------------------------------------
template <int CIN, int NK, int MODE>
__global__ __launch_bounds__(256)
void conv_kernel(const _Float16* __restrict__ Hr, const _Float16* __restrict__ Hi,
                 _Float16* __restrict__ Or, _Float16* __restrict__ Oi,
                 float* __restrict__ Y,
                 const _Float16* __restrict__ Bfrag, const float* __restrict__ bias)
{
    constexpr int CSTR  = 2 * CIN;              // halves per x column
    constexpr int KSPAN = 3 * CSTR;             // valid K per dy row
    constexpr int KP    = (KSPAN + 7) & ~7;     // padded K per dy row
    constexpr int XCOLS = 20;                   // 16 + wrap halo
    constexpr int RS    = XCOLS * CSTR;         // halves per dy row
    constexpr int MS    = 3 * RS;               // halves per batch row m

    const int bt   = blockIdx.x;                // batch tile (16 rows)
    const int cy   = blockIdx.y;                // output lattice row
    const int tid  = threadIdx.x;
    const int lane = tid & 31, wave = tid >> 5;

    __shared__ __align__(32) _Float16 sA[16 * MS + 64];
    __shared__ __align__(32) _Float16 sB[NK * 2 * 32 * 16];

    __builtin_prefetch(Bfrag, 0, 1);            // global_prefetch_b8

    // stage B fragments (dword copies)
    {
        constexpr int ndw = NK * 2 * 32 * 16 / 2;
        const uint32_t* src = (const uint32_t*)Bfrag;
        uint32_t* dst = (uint32_t*)sB;
        #pragma unroll 2
        for (int i = tid; i < ndw; i += 256) dst[i] = src[i];
    }
    // stage activations: 16 m x 3 dy x 20 x-cols x (p*CIN+c)
    {
        constexpr int tot = 16 * MS;
        for (int it = tid; it < tot; it += 256) {
            int pc = it % CSTR;  int t2 = it / CSTR;
            int xx = t2 % XCOLS; int t3 = t2 / XCOLS;
            int dy = t3 % 3;     int m  = t3 / 3;
            int p = pc / CIN, c = pc - p * CIN;
            int gcell = (((cy + dy) & (L_ - 1)) << 4) | (xx & 15);
            size_t gidx = (((size_t)(bt * 16 + m)) * NC_ + gcell) * 16 + c;
            sA[it] = p ? Hi[gidx] : Hr[gidx];
        }
    }
    __syncthreads();

    const int n     = lane & 15;
    const int khalf = lane >> 4;
    const int m     = lane & 15;                // A-matrix row = M
    const int mD    = khalf * 8;                // D row base
    const int kbase = 8 * khalf;

    float bre_s = 0.f, bim_s = 0.f;
    if (n < NF_) { bre_s = bias[2 * n]; bim_s = bias[2 * n + 1]; }

    #pragma unroll
    for (int cc = 0; cc < 2; ++cc) {
        const int xo = wave + 8 * cc;           // output x within row
        const int j  = (cy << 4) | xo;          // global output cell
        const int abase = m * MS + xo * CSTR;   // per-lane A base (halves)
        v8f acc_re = {}; v8f acc_im = {};

        #pragma unroll
        for (int s = 0; s < NK; ++s) {
            v16h a;
            if constexpr ((CSTR % 8) == 0) {
                // two contiguous aligned 8-half runs per k-step
                int k0 = s * 32 + kbase;
                int k1 = k0 + 16;
                int d0 = k0 / KP; d0 = d0 > 2 ? 2 : d0;
                int d1 = k1 / KP; d1 = d1 > 2 ? 2 : d1;
                v8h lo = *(const v8h*)&sA[abase + d0 * (RS - KP) + k0];
                v8h hi = *(const v8h*)&sA[abase + d1 * (RS - KP) + k1];
                #pragma unroll
                for (int q = 0; q < 8; ++q) { a[q] = lo[q]; a[q + 8] = hi[q]; }
            } else {
                // scalar fallback (CIN=2): constant-folded offsets
                #pragma unroll
                for (int i = 0; i < 16; ++i) {
                    int k = s * 32 + kbase + (i < 8 ? i : i + 8);
                    int d = k / KP; d = d > 2 ? 2 : d;
                    a[i] = sA[abase + d * (RS - KP) + k];
                }
            }
            v16h bre = *(const v16h*)&sB[((s * 2 + 0) * 32 + lane) * 16];
            v16h bim = *(const v16h*)&sB[((s * 2 + 1) * 32 + lane) * 16];
            acc_re = __builtin_amdgcn_wmma_f32_16x16x32_f16(
                false, a, false, bre, (short)0, acc_re, false, false);
            acc_im = __builtin_amdgcn_wmma_f32_16x16x32_f16(
                false, a, false, bim, (short)0, acc_im, false, false);
        }

        // epilogue: bias + mode
        #pragma unroll
        for (int rr = 0; rr < 8; ++rr) {
            float zr = acc_re[rr] + bre_s;
            float zi = acc_im[rr] + bim_s;
            int mm = rr + mD;
            size_t row = (size_t)(bt * 16 + mm);
            if (n < NF_) {
                if constexpr (MODE == 0) {
                    // act2(z) = z/2 + z^2/4
                    float ar = 0.5f * zr + 0.25f * (zr * zr - zi * zi);
                    float ai = 0.5f * zi + 0.5f  * (zr * zi);
                    size_t o = (row * NC_ + j) * 16 + n;
                    Or[o] = (_Float16)ar;  Oi[o] = (_Float16)ai;
                } else if constexpr (MODE == 1) {
                    size_t o = ((row * NC_ + j) * NF_ + n) * 2;
                    Y[o] = zr;  Y[o + 1] = zi;
                } else {
                    size_t o = ((row * NC_ + j) * NF_ + n) * 2;
                    Y[o] += zr; Y[o + 1] += zi;
                }
            }
        }
    }
}

// ---------------------------------------------------------------------------
// Kernel 4: act4 reduction + alpha terms + log-mean-exp over 8 rotations
// ---------------------------------------------------------------------------
__global__ __launch_bounds__(256)
void final_kernel(const float* __restrict__ Y, const float* __restrict__ outA,
                  float* out)
{
    const int b   = blockIdx.x;
    const int tid = threadIdx.x;
    __shared__ float rbr[256], rbi[256];
    __shared__ float zr8[NROT_], zi8[NROT_];
    const float inv_sqrt3 = 0.57735026918962576f;

    for (int rot = 0; rot < NROT_; ++rot) {
        size_t row = (size_t)b * NROT_ + rot;
        const float* Yr = Y + row * NC_ * NF_ * 2;
        float sr = 0.f, si = 0.f;
        for (int idx = tid; idx < NC_ * NF_; idx += 256) {
            float zr = Yr[2 * idx], zi = Yr[2 * idx + 1];
            float z2r = zr * zr - zi * zi, z2i = 2.f * zr * zi;
            float z4r = z2r * z2r - z2i * z2i, z4i = 2.f * z2r * z2i;
            sr += 0.5f * zr + 0.25f * z2r - z4r * (1.f / 48.f);
            si += 0.5f * zi + 0.25f * z2i - z4i * (1.f / 48.f);
        }
        rbr[tid] = sr; rbi[tid] = si;
        __syncthreads();
        for (int st = 128; st > 0; st >>= 1) {
            if (tid < st) { rbr[tid] += rbr[tid + st]; rbi[tid] += rbi[tid + st]; }
            __syncthreads();
        }
        if (tid == 0) {
            zr8[rot] = rbr[0] * inv_sqrt3 + outA[2 * row];
            zi8[rot] = rbi[0] * inv_sqrt3 + outA[2 * row + 1];
        }
        __syncthreads();
    }
    if (tid == 0) {
        float mr = 0.f, mi = 0.f;
        for (int k = 0; k < NROT_; ++k) {
            float e = expf(zr8[k]);
            mr += e * cosf(zi8[k]);
            mi += e * sinf(zi8[k]);
        }
        mr *= (1.f / NROT_); mi *= (1.f / NROT_);
        out[2 * b]     = 0.5f * logf(mr * mr + mi * mi);   // log|w|
        out[2 * b + 1] = atan2f(mi, mr);                   // arg(w)
    }
}

// ---------------------------------------------------------------------------
extern "C" void kernel_launch(void* const* d_in, const int* in_sizes, int n_in,
                              void* d_out, int out_size, void* d_ws, size_t ws_size,
                              hipStream_t stream)
{
    (void)in_sizes; (void)n_in; (void)out_size; (void)ws_size;

    const int*   pg    = (const int*)d_in[0];
    /* d_in[1] = kernel3 (taps derived analytically) */
    const int*   tsite = (const int*)d_in[2];
    const int*   tcell = (const int*)d_in[3];
    const int*   invm  = (const int*)d_in[4];
    const int*   tfm   = (const int*)d_in[5];
    const int*   ltri  = (const int*)d_in[6];
    const int*   rtri  = (const int*)d_in[7];
    const float* kx    = (const float*)d_in[8];
    const float* ky    = (const float*)d_in[9];
    const int*   x     = (const int*)d_in[10];
    const float* W1a   = (const float*)d_in[11];
    const float* W1b   = (const float*)d_in[12];
    const float* W1c   = (const float*)d_in[13];
    const float* W2a   = (const float*)d_in[14];
    const float* W2b   = (const float*)d_in[15];
    const float* W2c   = (const float*)d_in[16];
    const float* b1a   = (const float*)d_in[17];
    const float* b1b   = (const float*)d_in[18];
    const float* b1c   = (const float*)d_in[19];
    const float* b2a   = (const float*)d_in[20];
    const float* b2b   = (const float*)d_in[21];
    const float* b2c   = (const float*)d_in[22];
    const float* alpha0= (const float*)d_in[23];
    const float* alpha1= (const float*)d_in[24];

    char* ws = (char*)d_ws;
    _Float16* Ar = (_Float16*)(ws + OFF_AR);
    _Float16* Ai = (_Float16*)(ws + OFF_AI);
    _Float16* Br = (_Float16*)(ws + OFF_BR);
    _Float16* Bi = (_Float16*)(ws + OFF_BI);
    _Float16* Cr = (_Float16*)(ws + OFF_CR);
    _Float16* Ci = (_Float16*)(ws + OFF_CI);
    float*    Y  = (float*)(ws + OFF_Y);
    _Float16* BF = (_Float16*)(ws + OFF_BF);
    float*  outA = (float*)(ws + OFF_OUTA);

    prep_kernel<<<BR_, 256, 0, stream>>>(x, pg, tsite, tcell, invm, tfm, ltri, rtri,
                                         kx, ky, alpha0, alpha1, Ar, Ai, Cr, Ci, outA);

    wprep_kernel< 4, 3><<<1, 32, 0, stream>>>(W1a, BF + 0 * BFRAG_HALVES);
    wprep_kernel<12, 7><<<1, 32, 0, stream>>>(W1b, BF + 1 * BFRAG_HALVES);
    wprep_kernel<12, 7><<<1, 32, 0, stream>>>(W1c, BF + 2 * BFRAG_HALVES);
    wprep_kernel< 2, 2><<<1, 32, 0, stream>>>(W2a, BF + 3 * BFRAG_HALVES);
    wprep_kernel<12, 7><<<1, 32, 0, stream>>>(W2b, BF + 4 * BFRAG_HALVES);
    wprep_kernel<12, 7><<<1, 32, 0, stream>>>(W2c, BF + 5 * BFRAG_HALVES);

    dim3 grid(BR_ / 16, L_);
    // branch 1: u0 (Cin=4) -> 12 -> 12 -> Y (write)
    conv_kernel< 4, 3, 0><<<grid, 256, 0, stream>>>(Ar, Ai, Br, Bi, Y, BF + 0 * BFRAG_HALVES, b1a);
    conv_kernel<12, 7, 0><<<grid, 256, 0, stream>>>(Br, Bi, Ar, Ai, Y, BF + 1 * BFRAG_HALVES, b1b);
    conv_kernel<12, 7, 1><<<grid, 256, 0, stream>>>(Ar, Ai, Br, Bi, Y, BF + 2 * BFRAG_HALVES, b1c);
    // branch 2: u1 (Cin=2) -> 12 -> 12 -> Y (accumulate)
    conv_kernel< 2, 2, 0><<<grid, 256, 0, stream>>>(Cr, Ci, Br, Bi, Y, BF + 3 * BFRAG_HALVES, b2a);
    conv_kernel<12, 7, 0><<<grid, 256, 0, stream>>>(Br, Bi, Ar, Ai, Y, BF + 4 * BFRAG_HALVES, b2b);
    conv_kernel<12, 7, 2><<<grid, 256, 0, stream>>>(Ar, Ai, Br, Bi, Y, BF + 5 * BFRAG_HALVES, b2c);

    final_kernel<<<B_, 256, 0, stream>>>(Y, outA, (float*)d_out);
}
